// mLSTMCell_32581621908080
// MI455X (gfx1250) — matrix operand order repaired and checked by reading, not compiled, not run on hardware
//
#include <hip/hip_runtime.h>
#include <hip/hip_bf16.h>

// ---------------------------------------------------------------------------
// mLSTM parallel forward for MI455X (gfx1250), flash-style, bf16 WMMA,
// TDM (tensor_load_to_lds) double-buffered k/v tile staging.
// ---------------------------------------------------------------------------

typedef __attribute__((ext_vector_type(16))) __bf16 v16bf;
typedef __attribute__((ext_vector_type(8)))  float  v8f;
typedef __attribute__((ext_vector_type(4)))  unsigned int v4u;
typedef __attribute__((ext_vector_type(8)))  int v8i;
typedef __attribute__((ext_vector_type(4)))  int v4i;

union BF16Frag {
    v16bf v;
    uint4 u[2];
};

constexpr int SEQ = 2048;
constexpr int EMB = 1024;
constexpr int NHEAD = 8;
constexpr int DHEAD = 128;   // EMB / NHEAD
constexpr int BR = 128;      // query rows per block (8 waves x 16 rows)
constexpr int BC = 32;       // keys per inner tile

__device__ __forceinline__ unsigned short f2bf(float f) {
    unsigned int u = __float_as_uint(f);
    unsigned int r = u + 0x7FFFu + ((u >> 16) & 1u);
    return (unsigned short)(r >> 16);
}

// Issue a 2D TDM tile load: tile (tile0 x tile1) of bf16 elements from a
// row-major tensor with row stride stride0 (elements) into contiguous LDS.
// D# layout per CDNA5 ISA ch.8 (group0: count/lds/global/type,
// group1: data_size + dims + 48-bit stride). 2D => remaining groups zero.
__device__ __forceinline__ void tdm_load_2d(unsigned lds_off, const void* gptr,
                                            unsigned tile0, unsigned tile1,
                                            unsigned long long stride0) {
    unsigned long long ga = (unsigned long long)(uintptr_t)gptr;
    v4u g0;
    g0.x = 1u;                                            // count=1 (valid)
    g0.y = lds_off;                                       // lds_addr (bytes)
    g0.z = (unsigned)(ga & 0xFFFFFFFFull);                // global_addr[31:0]
    g0.w = (unsigned)((ga >> 32) & 0x01FFFFFFull) | (2u << 30); // [56:32]+type
    v8i g1;
    g1[0] = (int)(1u << 16);                              // data_size=1 (2B)
    g1[1] = (int)((tile0 & 0xFFFFu) << 16);               // tensor_dim0 lo
    g1[2] = (int)((tile0 >> 16) | ((tile1 & 0xFFFFu) << 16)); // td0 hi, td1 lo
    g1[3] = (int)((tile1 >> 16) | (tile0 << 16));         // td1 hi, tile_dim0
    g1[4] = (int)(tile1 & 0xFFFFu);                       // tile_dim1 (td2=0)
    g1[5] = (int)(unsigned)(stride0 & 0xFFFFFFFFull);     // dim0_stride lo
    g1[6] = (int)(unsigned)((stride0 >> 32) & 0xFFFFull); // dim0_stride hi
    g1[7] = 0;
    __builtin_amdgcn_tensor_load_to_lds(g0, g1, (v4i){0, 0, 0, 0},
                                        (v4i){0, 0, 0, 0},
                                        (v8i){0, 0, 0, 0, 0, 0, 0, 0}, 0);
}

// ---------------------------------------------------------------------------
// Kernel 1: gate projections. One block per sequence row.
// ---------------------------------------------------------------------------
__global__ __launch_bounds__(256) void gates_kernel(
    const float* __restrict__ q, const float* __restrict__ k,
    const float* __restrict__ v, const float* __restrict__ Wi,
    const float* __restrict__ bi, const float* __restrict__ Wf,
    const float* __restrict__ bfv, float* __restrict__ igbuf,
    float* __restrict__ logfbuf) {
    const int s = blockIdx.x;
    const int tid = threadIdx.x;
    float accI[NHEAD], accF[NHEAD];
#pragma unroll
    for (int h = 0; h < NHEAD; ++h) { accI[h] = 0.f; accF[h] = 0.f; }

    for (int e = tid; e < 3 * EMB; e += 256) {
        float x;
        if (e < EMB)            x = q[(size_t)s * EMB + e];
        else if (e < 2 * EMB)   x = k[(size_t)s * EMB + e - EMB];
        else                    x = v[(size_t)s * EMB + e - 2 * EMB];
#pragma unroll
        for (int h = 0; h < NHEAD; ++h) {
            accI[h] = fmaf(x, Wi[(size_t)h * 3 * EMB + e], accI[h]);
            accF[h] = fmaf(x, Wf[(size_t)h * 3 * EMB + e], accF[h]);
        }
    }
#pragma unroll
    for (int off = 16; off >= 1; off >>= 1) {
#pragma unroll
        for (int h = 0; h < NHEAD; ++h) {
            accI[h] += __shfl_xor(accI[h], off, 32);
            accF[h] += __shfl_xor(accF[h], off, 32);
        }
    }
    __shared__ float red[16][8];
    const int wave = tid >> 5, lane = tid & 31;
    if (lane == 0) {
#pragma unroll
        for (int h = 0; h < NHEAD; ++h) {
            red[h][wave] = accI[h];
            red[8 + h][wave] = accF[h];
        }
    }
    __syncthreads();
    if (tid < 16) {
        float sum = 0.f;
#pragma unroll
        for (int w = 0; w < 8; ++w) sum += red[tid][w];
        if (tid < 8) {
            igbuf[tid * SEQ + s] = sum + bi[tid];
        } else {
            const int h = tid - 8;
            float x = sum + bfv[h];
            logfbuf[h * SEQ + s] = fminf(x, 0.f) - log1pf(expf(-fabsf(x)));
        }
    }
}

// ---------------------------------------------------------------------------
// Kernel 2: inclusive cumsum of log-forget per head.
// ---------------------------------------------------------------------------
__global__ void scan_kernel(const float* __restrict__ logf,
                            float* __restrict__ F) {
    const int h = threadIdx.x;
    if (h >= NHEAD) return;
    float run = 0.f;
    for (int s = 0; s < SEQ; ++s) {
        run += logf[h * SEQ + s];
        F[h * SEQ + s] = run;
    }
}

// ---------------------------------------------------------------------------
// Kernel 3: convert q,k to bf16.
// ---------------------------------------------------------------------------
__global__ __launch_bounds__(256) void cvt_kernel(
    const float* __restrict__ q, const float* __restrict__ k,
    unsigned short* __restrict__ qb, unsigned short* __restrict__ kb) {
    const size_t i = ((size_t)blockIdx.x * 256 + threadIdx.x) * 4;
    float4 a = *(const float4*)(q + i);
    float4 b = *(const float4*)(k + i);
    uint2 oa, ob;
    oa.x = (unsigned)f2bf(a.x) | ((unsigned)f2bf(a.y) << 16);
    oa.y = (unsigned)f2bf(a.z) | ((unsigned)f2bf(a.w) << 16);
    ob.x = (unsigned)f2bf(b.x) | ((unsigned)f2bf(b.y) << 16);
    ob.y = (unsigned)f2bf(b.z) | ((unsigned)f2bf(b.w) << 16);
    *(uint2*)(qb + i) = oa;
    *(uint2*)(kb + i) = ob;
}

// ---------------------------------------------------------------------------
// Kernel 4: v -> per-head transposed bf16: vtb[h][d][s].
// ---------------------------------------------------------------------------
__global__ __launch_bounds__(256) void vtr_kernel(
    const float* __restrict__ v, unsigned short* __restrict__ vtb) {
    __shared__ float tile[32][33];
    const int e0 = blockIdx.x * 32;
    const int s0 = blockIdx.y * 32;
    for (int r = threadIdx.y; r < 32; r += 8)
        tile[threadIdx.x][r] = v[(size_t)(s0 + r) * EMB + e0 + threadIdx.x];
    __syncthreads();
    const int h = e0 / DHEAD;
    const int dbase = e0 - h * DHEAD;
    for (int r = threadIdx.y; r < 32; r += 8)
        vtb[(size_t)h * DHEAD * SEQ + (size_t)(dbase + r) * SEQ + s0 + threadIdx.x] =
            f2bf(tile[r][threadIdx.x]);
}

// ---------------------------------------------------------------------------
// Kernel 5: flash-style mLSTM core with TDM double-buffered tile staging.
// Grid: (SEQ/BR, NHEAD). Block: 256 (8 waves x 16 query rows).
// ---------------------------------------------------------------------------
__global__ __launch_bounds__(256) void mlstm_flash_kernel(
    const unsigned short* __restrict__ qb, const unsigned short* __restrict__ kb,
    const unsigned short* __restrict__ vtb, const float* __restrict__ igbuf,
    const float* __restrict__ Fbuf, float* __restrict__ hbuf) {
    __shared__ alignas(16) unsigned short kLds[2][BC * DHEAD];   // 2 x 8 KB
    __shared__ alignas(16) unsigned short vLds[2][DHEAD * BC];   // 2 x 8 KB
    __shared__ alignas(16) unsigned short pLds[8][16 * BC];      // 8 KB

    const int h    = blockIdx.y;
    const int r0   = blockIdx.x * BR;
    const int tid  = threadIdx.x;
    const int wave = tid >> 5;
    const int lane = tid & 31;
    const int half = lane >> 4;
    const int ln   = lane & 15;
    const int rw   = r0 + wave * 16;

    // q A-fragments (A layout: lanes0-15 K{0..7,16..23}, lanes16-31 K{8..15,24..31})
    BF16Frag aq[4];
    const unsigned short* qrow = qb + (size_t)(rw + ln) * EMB + h * DHEAD;
#pragma unroll
    for (int kk = 0; kk < 4; ++kk) {
        aq[kk].u[0] = *(const uint4*)(qrow + kk * 32 + half * 8);
        aq[kk].u[1] = *(const uint4*)(qrow + kk * 32 + 16 + half * 8);
    }

    float fs[8];
#pragma unroll
    for (int j = 0; j < 8; ++j)
        fs[j] = Fbuf[h * SEQ + rw + j + half * 8];

    float m8[8], b8[8];
    v8f acc[8];
#pragma unroll
    for (int j = 0; j < 8; ++j) { m8[j] = -3.0e38f; b8[j] = 0.f; }
#pragma unroll
    for (int dt = 0; dt < 8; ++dt) acc[dt] = (v8f){};

    const float invSqrtD = 0.08838834764831845f;  // 1/sqrt(128)
    const int tEnd = r0 + BR;

    // TDM issue of one (k, vT) tile pair into buffer ib. Wave 0 only:
    // TDM is a per-wave DMA op (EXEC-independent), one wave must own it.
    auto issue_tile = [&](int t0, int ib) {
        tdm_load_2d((unsigned)(uintptr_t)&kLds[ib][0],
                    kb + (size_t)t0 * EMB + h * DHEAD,
                    DHEAD, BC, (unsigned long long)EMB);
        tdm_load_2d((unsigned)(uintptr_t)&vLds[ib][0],
                    vtb + (size_t)h * DHEAD * SEQ + t0,
                    BC, DHEAD, (unsigned long long)SEQ);
    };

    if (wave == 0) {
        issue_tile(0, 0);
        __builtin_amdgcn_s_wait_tensorcnt(0);
    }
    __syncthreads();

    int ib = 0;
    for (int t0 = 0; t0 < tEnd; t0 += BC, ib ^= 1) {
        // prefetch next tile pair into the spare buffer (overlaps compute)
        if (wave == 0 && (t0 + BC) < tEnd) issue_tile(t0 + BC, ib ^ 1);

        if (t0 <= rw + 15) {   // tile not fully masked for this wave
            const unsigned short* kT = &kLds[ib][0];
            const unsigned short* vT = &vLds[ib][0];

            // ---- S = q . k^T : two 16x16 tiles, K=128 in 4 WMMA steps ----
            v8f sc0 = (v8f){}, sc1 = (v8f){};
#pragma unroll
            for (int kk = 0; kk < 4; ++kk) {
                // B layout: lane=N(key); same K order as A
                // (lanes0-15 -> K{0..7,16..23}, lanes16-31 -> K{8..15,24..31}).
                BF16Frag bk0, bk1;
                const int cofs = kk * 32 + half * 8;
                bk0.u[0] = *(const uint4*)(&kT[ln * DHEAD + cofs]);
                bk0.u[1] = *(const uint4*)(&kT[ln * DHEAD + cofs + 16]);
                bk1.u[0] = *(const uint4*)(&kT[(16 + ln) * DHEAD + cofs]);
                bk1.u[1] = *(const uint4*)(&kT[(16 + ln) * DHEAD + cofs + 16]);
                sc0 = __builtin_amdgcn_wmma_f32_16x16x32_bf16(
                    false, aq[kk].v, false, bk0.v, (short)0, sc0, false, false);
                sc1 = __builtin_amdgcn_wmma_f32_16x16x32_bf16(
                    false, aq[kk].v, false, bk1.v, (short)0, sc1, false, false);
            }

            // ---- decay + online stats (C layout: lane=N, vgpr j=M) ----
            const int t_a = t0 + ln;
            const int t_b = t0 + 16 + ln;
            const float ga = igbuf[h * SEQ + t_a] - Fbuf[h * SEQ + t_a];
            const float gb = igbuf[h * SEQ + t_b] - Fbuf[h * SEQ + t_b];

            float logD0[8], logD1[8], mnew[8], scale[8];
#pragma unroll
            for (int j = 0; j < 8; ++j) {
                const int srow = rw + j + half * 8;
                float d0 = fs[j] + ga; if (t_a > srow) d0 = -3.0e38f;
                float d1 = fs[j] + gb; if (t_b > srow) d1 = -3.0e38f;
                logD0[j] = d0; logD1[j] = d1;
                float cand = fmaxf(d0, d1);
#pragma unroll
                for (int off = 8; off >= 1; off >>= 1)
                    cand = fmaxf(cand, __shfl_xor(cand, off, 16));
                const float mo = m8[j];
                const float mn = fmaxf(mo, cand);
                mnew[j] = mn;
                scale[j] = __expf(mo - mn);
                m8[j] = mn;
            }

            float p0[8], p1[8];
#pragma unroll
            for (int j = 0; j < 8; ++j) {
                const int srow = rw + j + half * 8;
                const float e0v = (t_a <= srow) ? __expf(logD0[j] - mnew[j]) : 0.f;
                const float e1v = (t_b <= srow) ? __expf(logD1[j] - mnew[j]) : 0.f;
                p0[j] = sc0[j] * invSqrtD * e0v;
                p1[j] = sc1[j] * invSqrtD * e1v;
                float rs = p0[j] + p1[j];
#pragma unroll
                for (int off = 8; off >= 1; off >>= 1)
                    rs += __shfl_xor(rs, off, 16);
                b8[j] = b8[j] * scale[j] + rs;
#pragma unroll
                for (int dt = 0; dt < 8; ++dt) acc[dt][j] *= scale[j];
            }

            // ---- C-layout -> A-layout for P via per-wave LDS scratch ----
#pragma unroll
            for (int j = 0; j < 8; ++j) {
                const int rl = j + half * 8;
                pLds[wave][rl * BC + ln]      = f2bf(p0[j]);
                pLds[wave][rl * BC + 16 + ln] = f2bf(p1[j]);
            }
            asm volatile("s_wait_dscnt 0x0" ::: "memory");
            BF16Frag pa;
            pa.u[0] = *(const uint4*)(&pLds[wave][ln * BC + half * 8]);
            pa.u[1] = *(const uint4*)(&pLds[wave][ln * BC + 16 + half * 8]);

            // ---- acc += P @ V ----
#pragma unroll
            for (int dt = 0; dt < 8; ++dt) {
                BF16Frag bv;
                bv.u[0] = *(const uint4*)(&vT[(dt * 16 + ln) * BC + half * 8]);
                bv.u[1] = *(const uint4*)(&vT[(dt * 16 + ln) * BC + 16 + half * 8]);
                acc[dt] = __builtin_amdgcn_wmma_f32_16x16x32_bf16(
                    false, pa.v, false, bv.v, (short)0, acc[dt], false, false);
            }
        }

        // DMA for the next tile must land before anyone reads the spare buf.
        if (wave == 0) __builtin_amdgcn_s_wait_tensorcnt(0);
        __syncthreads();
    }

    // ---- epilogue: mLSTM normalizer, write h (f32) ----
    float inv[8];
#pragma unroll
    for (int j = 0; j < 8; ++j) {
        const float nrm = fmaxf(fabsf(b8[j]), __expf(-m8[j]));
        inv[j] = 1.0f / (nrm + 1e-6f);
    }
#pragma unroll
    for (int dt = 0; dt < 8; ++dt) {
#pragma unroll
        for (int j = 0; j < 8; ++j) {
            const int row = rw + j + half * 8;
            const int col = h * DHEAD + dt * 16 + ln;
            hbuf[(size_t)row * EMB + col] = acc[dt][j] * inv[j];
        }
    }
}

// ---------------------------------------------------------------------------
// Kernel 6: in-place residual LayerNorm: out = LN(h) * (1 + w).
// ---------------------------------------------------------------------------
__global__ __launch_bounds__(256) void ln_kernel(float* __restrict__ hio,
                                                 const float* __restrict__ w) {
    const int s = blockIdx.x;
    const int tid = threadIdx.x;
    float x[4];
    float sum = 0.f, sumsq = 0.f;
#pragma unroll
    for (int i = 0; i < 4; ++i) {
        x[i] = hio[(size_t)s * EMB + tid + i * 256];
        sum += x[i];
        sumsq += x[i] * x[i];
    }
#pragma unroll
    for (int off = 16; off >= 1; off >>= 1) {
        sum += __shfl_xor(sum, off, 32);
        sumsq += __shfl_xor(sumsq, off, 32);
    }
    __shared__ float sA[8], sB[8];
    const int wave = tid >> 5, lane = tid & 31;
    if (lane == 0) { sA[wave] = sum; sB[wave] = sumsq; }
    __syncthreads();
    if (tid == 0) {
        float ts = 0.f, tq = 0.f;
        for (int wv = 0; wv < 8; ++wv) { ts += sA[wv]; tq += sB[wv]; }
        const float mu = ts / EMB;
        sA[0] = mu;
        sB[0] = tq / EMB - mu * mu;
    }
    __syncthreads();
    const float mu = sA[0];
    const float rstd = rsqrtf(sB[0] + 1e-5f);
#pragma unroll
    for (int i = 0; i < 4; ++i) {
        const int e = tid + i * 256;
        hio[(size_t)s * EMB + e] = (x[i] - mu) * rstd * (1.f + w[e]);
    }
}

// ---------------------------------------------------------------------------
extern "C" void kernel_launch(void* const* d_in, const int* in_sizes, int n_in,
                              void* d_out, int out_size, void* d_ws,
                              size_t ws_size, hipStream_t stream) {
    const float* q   = (const float*)d_in[0];
    const float* k   = (const float*)d_in[1];
    const float* v   = (const float*)d_in[2];
    const float* Wi  = (const float*)d_in[3];
    const float* bi  = (const float*)d_in[4];
    const float* Wf  = (const float*)d_in[5];
    const float* bfv = (const float*)d_in[6];
    const float* lnw = (const float*)d_in[7];
    float* out = (float*)d_out;

    char* ws = (char*)d_ws;
    unsigned short* qbW  = (unsigned short*)ws;                           // 4 MB
    unsigned short* kbW  = (unsigned short*)(ws + (size_t)SEQ * EMB * 2); // 4 MB
    unsigned short* vtbW = (unsigned short*)(ws + (size_t)SEQ * EMB * 4); // 4 MB
    float* igbuf   = (float*)(ws + (size_t)SEQ * EMB * 6);                // 64 KB
    float* logfbuf = igbuf + NHEAD * SEQ;                                 // 64 KB
    float* Fbuf    = logfbuf + NHEAD * SEQ;                               // 64 KB

    gates_kernel<<<dim3(SEQ), dim3(256), 0, stream>>>(q, k, v, Wi, bi, Wf, bfv,
                                                      igbuf, logfbuf);
    scan_kernel<<<dim3(1), dim3(32), 0, stream>>>(logfbuf, Fbuf);
    cvt_kernel<<<dim3((SEQ * EMB) / (256 * 4)), dim3(256), 0, stream>>>(q, k,
                                                                        qbW, kbW);
    vtr_kernel<<<dim3(EMB / 32, SEQ / 32), dim3(32, 8), 0, stream>>>(v, vtbW);
    mlstm_flash_kernel<<<dim3(SEQ / BR, NHEAD), dim3(256), 0, stream>>>(
        qbW, kbW, vtbW, igbuf, Fbuf, out);
    ln_kernel<<<dim3(SEQ), dim3(256), 0, stream>>>(out, lnw);
}
